// DistNN_88794153877510
// MI455X (gfx1250) — matrix operand order repaired and checked by reading, not compiled, not run on hardware
//
#include <hip/hip_runtime.h>
#include <hip/hip_bf16.h>

typedef __bf16 bf16;
typedef __attribute__((ext_vector_type(16))) __bf16 v16bf;
typedef __attribute__((ext_vector_type(8)))  __bf16 v8bf;
typedef __attribute__((ext_vector_type(8)))  float   v8f;
typedef __attribute__((ext_vector_type(4)))  float   v4f;

#define LDW1 136   // padded row stride (halves) for 128-K weights
#define LDW2 392   // padded row stride (halves) for 384-K weights
#define LDI  136   // padded row stride (halves) for intermediate tile

__device__ __forceinline__ v8f wmma_bf16(v16bf a, v16bf b, v8f c) {
  return __builtin_amdgcn_wmma_f32_16x16x32_bf16(false, a, false, b, (short)0, c, false, false);
}

// A fragment (16x32 bf16) from global fp32, streamed non-temporally (preserve L2 for h).
// Caller passes p = &src[row*ld + kbase + g*8].
// Lane layout: lane holds row, halves 0..7 -> K = kbase+g*8+i, halves 8..15 -> K = kbase+16+g*8+i.
__device__ __forceinline__ v16bf load_a_global(const float* __restrict__ p) {
  v4f a0 = __builtin_nontemporal_load((const v4f*)(p));
  v4f a1 = __builtin_nontemporal_load((const v4f*)(p + 4));
  v4f a2 = __builtin_nontemporal_load((const v4f*)(p + 16));
  v4f a3 = __builtin_nontemporal_load((const v4f*)(p + 20));
  v16bf a;
#pragma unroll
  for (int i = 0; i < 4; i++) {
    a[i]      = (bf16)a0[i];
    a[4 + i]  = (bf16)a1[i];
    a[8 + i]  = (bf16)a2[i];
    a[12 + i] = (bf16)a3[i];
  }
  return a;
}

// A fragment from LDS bf16 tile. q = &ibuf[row*LDI + kbase + g*8]; runs at +0 and +16.
__device__ __forceinline__ v16bf load_a_lds(const bf16* q) {
  v8bf lo = *(const v8bf*)(q);
  v8bf hi = *(const v8bf*)(q + 16);
  v16bf a;
#pragma unroll
  for (int i = 0; i < 8; i++) { a[i] = lo[i]; a[8 + i] = hi[i]; }
  return a;
}

// B fragment (32x16 bf16). Weights stored transposed [N][K] (padded). Lane holds col N,
// halves 0..7 -> K = kbase+g*8+i, halves 8..15 -> K = kbase+16+g*8+i.
__device__ __forceinline__ v16bf load_b_lds(const bf16* q) {
  v8bf lo = *(const v8bf*)(q);
  v8bf hi = *(const v8bf*)(q + 16);
  v16bf b;
#pragma unroll
  for (int i = 0; i < 8; i++) { b[i] = lo[i]; b[8 + i] = hi[i]; }
  return b;
}

__device__ __forceinline__ void copy_w_lds(bf16* dst, const bf16* __restrict__ src,
                                           int K, int ldd, int tid) {
  const int nvec = 128 * K / 8;
  for (int i = tid; i < nvec; i += 256) {
    int row = i / (K / 8);
    int cv  = i - row * (K / 8);
    v8bf v = *(const v8bf*)(src + row * K + cv * 8);
    *(v8bf*)(dst + row * ldd + cv * 8) = v;
  }
}

// ---------------- prep: transpose weights to bf16 [N][K], zero stats/pool -------------
__global__ void prep_kernel(const float* __restrict__ W_ea, const float* __restrict__ W_er,
                            const float* __restrict__ W_eb, const float* __restrict__ W_fr,
                            const float* __restrict__ W_fb,
                            bf16* Wt_ea, bf16* Wt_er, bf16* Wt_eb, bf16* Wt_fr, bf16* Wt_fb,
                            float* zero_ptr, int nzero) {
  int tid = blockIdx.x * blockDim.x + threadIdx.x;
  int stride = gridDim.x * blockDim.x;
  for (int i = tid; i < 16384; i += stride) {
    int n = i & 127, k = i >> 7;
    Wt_ea[n * 128 + k] = (bf16)W_ea[k * 128 + n];
    Wt_er[n * 128 + k] = (bf16)W_er[k * 128 + n];
    Wt_eb[n * 128 + k] = (bf16)W_eb[k * 128 + n];
  }
  for (int i = tid; i < 49152; i += stride) {
    int n = i & 127, k = i >> 7;           // k in 0..383
    Wt_fr[n * 384 + k] = (bf16)W_fr[k * 128 + n];
    Wt_fb[n * 384 + k] = (bf16)W_fb[k * 128 + n];
  }
  for (int i = tid; i < nzero; i += stride) zero_ptr[i] = 0.0f;
}

// ---------------- fused edge kernel --------------------------------------------------
__global__ __launch_bounds__(256)
void fused_edge_kernel(const float* __restrict__ atom, const float* __restrict__ rdf,
                       const float* __restrict__ bdf,
                       const float* __restrict__ b_ea, const float* __restrict__ b_er,
                       const float* __restrict__ b_eb, const float* __restrict__ b_fr,
                       const float* __restrict__ b_fb,
                       const bf16* __restrict__ Wt_ea, const bf16* __restrict__ Wt_er,
                       const bf16* __restrict__ Wt_eb, const bf16* __restrict__ Wt_fr,
                       const bf16* __restrict__ Wt_fb,
                       bf16* __restrict__ hout, float* gsum, float* gsq, int ntiles) {
  __shared__ bf16 sWea[128 * LDW1];
  __shared__ bf16 sWer[128 * LDW1];
  __shared__ bf16 sWeb[128 * LDW1];
  __shared__ bf16 sWfr[128 * LDW2];
  __shared__ bf16 sWfb[128 * LDW2];
  __shared__ bf16 ibuf[64 * LDI];

  const int tid  = threadIdx.x;
  copy_w_lds(sWea, Wt_ea, 128, LDW1, tid);
  copy_w_lds(sWer, Wt_er, 128, LDW1, tid);
  copy_w_lds(sWeb, Wt_eb, 128, LDW1, tid);
  copy_w_lds(sWfr, Wt_fr, 384, LDW2, tid);
  copy_w_lds(sWfb, Wt_fb, 384, LDW2, tid);
  __syncthreads();

  const int wave = tid >> 5;
  const int lane = tid & 31;
  const int l16  = lane & 15;
  const int g    = lane >> 4;
  const int mrow = (wave & 3) * 16;     // 4 M-strips of 16 rows
  const int nh   = wave >> 2;           // 2 N-halves of 64 cols

  float ssr[4] = {0, 0, 0, 0}, sqr[4] = {0, 0, 0, 0};
  float ssb[4] = {0, 0, 0, 0}, sqb[4] = {0, 0, 0, 0};

  for (int tile = blockIdx.x; tile < ntiles; tile += gridDim.x) {
    const long e0 = (long)tile * 64;
    v8f hr[4] = {}, hb[4] = {};

    // ---- 4 phases: atom1, atom2, x_rdf, x_bdf --------------------------------------
#pragma unroll 1
    for (int phase = 0; phase < 4; phase++) {
      const float* src;
      int ld, col0, act;
      const bf16* sW;
      const float* bias;
      if (phase == 0)      { src = atom; ld = 256; col0 = 0;   sW = sWea; bias = b_ea; act = 0; }
      else if (phase == 1) { src = atom; ld = 256; col0 = 128; sW = sWea; bias = b_ea; act = 0; }
      else if (phase == 2) { src = rdf;  ld = 128; col0 = 0;   sW = sWer; bias = b_er; act = 1; }
      else                 { src = bdf;  ld = 128; col0 = 0;   sW = sWeb; bias = b_eb; act = 1; }

      // stage 1: [64 x 128] = act(X @ W + b)
      v8f c[4] = {};
      const float* rowp = src + (e0 + mrow + l16) * (long)ld + col0 + g * 8;
#pragma unroll
      for (int kb = 0; kb < 128; kb += 32) {
        v16bf a = load_a_global(rowp + kb);
#pragma unroll
        for (int nt = 0; nt < 4; nt++) {
          v16bf b = load_b_lds(sW + (nh * 64 + nt * 16 + l16) * LDW1 + kb + g * 8);
          c[nt] = wmma_bf16(a, b, c[nt]);
        }
      }
#pragma unroll
      for (int nt = 0; nt < 4; nt++) {
        const int col = nh * 64 + nt * 16 + l16;
        const float bv = bias[col];
#pragma unroll
        for (int j = 0; j < 8; j++) {
          float v = c[nt][j] + bv;
          v = fmaxf(v, 0.0f);
          if (act == 1) v = fminf(v, 6.0f);
          ibuf[(mrow + 8 * g + j) * LDI + col] = (bf16)v;
        }
      }
      __syncthreads();

      // stage 2: contribute K-slice [phase_k*128 .. +128) of W_fr / W_fb
      const int kofs = (phase < 2 ? phase : 2) * 128;
      const bool do_r = (phase != 3);
      const bool do_b = (phase != 2);
#pragma unroll
      for (int kb = 0; kb < 128; kb += 32) {
        v16bf a = load_a_lds(ibuf + (mrow + l16) * LDI + kb + g * 8);
#pragma unroll
        for (int nt = 0; nt < 4; nt++) {
          const int nrow = (nh * 64 + nt * 16 + l16);
          if (do_r) {
            v16bf b = load_b_lds(sWfr + nrow * LDW2 + kofs + kb + g * 8);
            hr[nt] = wmma_bf16(a, b, hr[nt]);
          }
          if (do_b) {
            v16bf b = load_b_lds(sWfb + nrow * LDW2 + kofs + kb + g * 8);
            hb[nt] = wmma_bf16(a, b, hb[nt]);
          }
        }
      }
      __syncthreads();
    }

    // ---- epilogue: bias, write h (bf16, default TH -> stays in 192MB L2 for pool pass),
    //      accumulate BN stats in registers ------------------------------------------
#pragma unroll
    for (int nt = 0; nt < 4; nt++) {
      const int col = nh * 64 + nt * 16 + l16;
      const float bvr = b_fr[col];
      const float bvb = b_fb[col];
#pragma unroll
      for (int j = 0; j < 8; j++) {
        const long e = e0 + mrow + 8 * g + j;
        float vr = hr[nt][j] + bvr;
        float vb = hb[nt][j] + bvb;
        hout[e * 256 + col]       = (bf16)vr;
        hout[e * 256 + col + 128] = (bf16)vb;
        ssr[nt] += vr; sqr[nt] += vr * vr;
        ssb[nt] += vb; sqb[nt] += vb * vb;
      }
    }
  }

  // one atomic flush per block (combine lane halves first)
#pragma unroll
  for (int nt = 0; nt < 4; nt++) {
    float sr = ssr[nt] + __shfl_down(ssr[nt], 16, 32);
    float qr = sqr[nt] + __shfl_down(sqr[nt], 16, 32);
    float sb = ssb[nt] + __shfl_down(ssb[nt], 16, 32);
    float qb = sqb[nt] + __shfl_down(sqb[nt], 16, 32);
    if (g == 0) {
      const int col = nh * 64 + nt * 16 + l16;
      atomicAdd(&gsum[col], sr);       atomicAdd(&gsq[col], qr);
      atomicAdd(&gsum[col + 128], sb); atomicAdd(&gsq[col + 128], qb);
    }
  }
}

// ---------------- BN finalize --------------------------------------------------------
__global__ void bn_finalize_kernel(const float* __restrict__ gsum, const float* __restrict__ gsq,
                                   const float* __restrict__ gamma, const float* __restrict__ beta,
                                   float* scale, float* shiftv, float invE) {
  int c = threadIdx.x;  // 256 threads
  float mu = gsum[c] * invE;
  float var = gsq[c] * invE - mu * mu;
  float sc = gamma[c] * rsqrtf(var + 1e-5f);
  scale[c] = sc;
  shiftv[c] = beta[c] - mu * sc;
}

// ---------------- normalize + relu + segment mean-pool (sum part) --------------------
#define POOL_CHUNK 512
__global__ __launch_bounds__(256)
void pool_kernel(const bf16* __restrict__ h, const int* __restrict__ gidx,
                 const float* __restrict__ scale, const float* __restrict__ shiftv,
                 float* pooled, float* cnt, int E) {
  __shared__ int sg[POOL_CHUNK];
  const long e0 = (long)blockIdx.x * POOL_CHUNK;
  const int t = threadIdx.x;
  const int nloc = (int)min((long)POOL_CHUNK, (long)E - e0);
  for (int i = t; i < nloc; i += 256) sg[i] = gidx[e0 + i];
  __syncthreads();
  const float sc = scale[t], sh = shiftv[t];
  float acc = 0.0f, cc = 0.0f;
  int cur = sg[0];
  for (int i = 0; i < nloc; i++) {
    int gg = sg[i];
    if (gg != cur) {
      atomicAdd(&pooled[(long)cur * 256 + t], acc);
      if (t == 0) atomicAdd(&cnt[cur], cc);
      acc = 0.0f; cc = 0.0f; cur = gg;
    }
    float v = (float)h[(e0 + i) * 256 + t] * sc + sh;
    v = fmaxf(v, 0.0f);
    acc += v; cc += 1.0f;
  }
  atomicAdd(&pooled[(long)cur * 256 + t], acc);
  if (t == 0) atomicAdd(&cnt[cur], cc);
}

// ---------------- head MLP -----------------------------------------------------------
__global__ void head_kernel(const float* __restrict__ pooled, const float* __restrict__ cnt,
                            const float* __restrict__ ref, const float* __restrict__ W1,
                            const float* __restrict__ b1, const float* __restrict__ W2,
                            const float* __restrict__ b2, float* out, int G) {
  int gidx = blockIdx.x * blockDim.x + threadIdx.x;
  if (gidx >= G) return;
  float inv = 1.0f / fmaxf(cnt[gidx], 1.0f);
  float z[16];
#pragma unroll
  for (int j = 0; j < 16; j++) z[j] = b1[j];
  for (int k = 0; k < 256; k++) {
    float x = pooled[(long)gidx * 256 + k] * inv;
#pragma unroll
    for (int j = 0; j < 16; j++) z[j] += x * W1[k * 16 + j];
  }
  float xr = ref[gidx];
#pragma unroll
  for (int j = 0; j < 16; j++) z[j] += xr * W1[256 * 16 + j];
  float o = b2[0];
#pragma unroll
  for (int j = 0; j < 16; j++) o += fmaxf(z[j], 0.0f) * W2[j];
  out[gidx] = fmaxf(o, 0.0f);
}

// ---------------- launcher -----------------------------------------------------------
extern "C" void kernel_launch(void* const* d_in, const int* in_sizes, int n_in,
                              void* d_out, int out_size, void* d_ws, size_t ws_size,
                              hipStream_t stream) {
  const float* atom_feat = (const float*)d_in[0];
  const float* rdf_feat  = (const float*)d_in[1];
  const float* bdf_feat  = (const float*)d_in[2];
  const int*   graph_idx = (const int*)d_in[6];
  const float* ref_feat  = (const float*)d_in[7];
  const float* W_ea = (const float*)d_in[8];   const float* b_ea = (const float*)d_in[9];
  const float* W_er = (const float*)d_in[10];  const float* b_er = (const float*)d_in[11];
  const float* W_eb = (const float*)d_in[12];  const float* b_eb = (const float*)d_in[13];
  const float* W_fr = (const float*)d_in[14];  const float* b_fr = (const float*)d_in[15];
  const float* W_fb = (const float*)d_in[16];  const float* b_fb = (const float*)d_in[17];
  const float* bn_gamma = (const float*)d_in[18];
  const float* bn_beta  = (const float*)d_in[19];
  const float* W1 = (const float*)d_in[20];    const float* b1 = (const float*)d_in[21];
  const float* W2 = (const float*)d_in[22];    const float* b2 = (const float*)d_in[23];

  const int E = in_sizes[1] / 128;
  const int G = in_sizes[7];

  char* ws = (char*)d_ws;
  bf16*  Wt_ea  = (bf16*)(ws);
  bf16*  Wt_er  = (bf16*)(ws + 32768);
  bf16*  Wt_eb  = (bf16*)(ws + 65536);
  bf16*  Wt_fr  = (bf16*)(ws + 98304);
  bf16*  Wt_fb  = (bf16*)(ws + 196608);
  float* gsum   = (float*)(ws + 294912);
  float* gsq    = (float*)(ws + 295936);
  float* scale  = (float*)(ws + 296960);
  float* shiftv = (float*)(ws + 297984);
  float* cnt    = (float*)(ws + 299008);
  float* pooled = (float*)(ws + 303104);
  bf16*  hbuf   = (bf16*)(ws + 303104 + (size_t)G * 256 * 4);

  // zero: gsum..end of pooled  (gsum,gsq,scale,shift,cnt,pooled — contiguous)
  const int nzero = (int)((303104 + (size_t)G * 256 * 4 - 294912) / 4);

  prep_kernel<<<1024, 256, 0, stream>>>(W_ea, W_er, W_eb, W_fr, W_fb,
                                        Wt_ea, Wt_er, Wt_eb, Wt_fr, Wt_fb,
                                        gsum, nzero);

  const int ntiles = E / 64;
  int nblocks = ntiles < 1024 ? ntiles : 1024;
  fused_edge_kernel<<<nblocks, 256, 0, stream>>>(atom_feat, rdf_feat, bdf_feat,
                                                 b_ea, b_er, b_eb, b_fr, b_fb,
                                                 Wt_ea, Wt_er, Wt_eb, Wt_fr, Wt_fb,
                                                 hbuf, gsum, gsq, ntiles);

  bn_finalize_kernel<<<1, 256, 0, stream>>>(gsum, gsq, bn_gamma, bn_beta,
                                            scale, shiftv, 1.0f / (float)E);

  const int pblocks = (E + POOL_CHUNK - 1) / POOL_CHUNK;
  pool_kernel<<<pblocks, 256, 0, stream>>>(hbuf, graph_idx, scale, shiftv, pooled, cnt, E);

  head_kernel<<<(G + 255) / 256, 256, 0, stream>>>(pooled, cnt, ref_feat, W1, b1, W2, b2,
                                                   (float*)d_out, G);
}
